// LocalWindowAttention_27281632264486
// MI455X (gfx1250) — matrix operand order repaired and checked by reading, not compiled, not run on hardware
//
#include <hip/hip_runtime.h>

// ---------------- problem constants ----------------
#define EMB    1024
#define HEADS  16
#define DHEAD  64
#define SEQ    4000
#define SEQP   4096
#define NWIN   32
#define BATCH  8
#define MROWS  (BATCH * SEQP)          // 32768
#define NQKV   (3 * EMB)               // 3072

typedef __bf16 bf16_t;
typedef bf16_t v16bf __attribute__((ext_vector_type(16)));
typedef float  v8f   __attribute__((ext_vector_type(8)));

// fp32 -> bf16, round to nearest (ties away): 2 VALU, no subregister moves
__device__ __forceinline__ unsigned short f2bf(float f) {
    return (unsigned short)((__float_as_uint(f) + 0x8000u) >> 16);
}

// two fp32 -> packed bf16 dword via v_perm_b32 (3 VALU per pair)
__device__ __forceinline__ unsigned pk_bf16(float a, float b) {
    const unsigned ua = __float_as_uint(a) + 0x8000u;
    const unsigned ub = __float_as_uint(b) + 0x8000u;
    // dst bytes {b[3],b[2],a[3],a[2]}  (S1 = ua -> sel 0-3, S0 = ub -> sel 4-7)
    return __builtin_amdgcn_perm(ub, ua, 0x07060302u);
}

// pack 8 fp32 (two float4) into 4 bf16 dwords, masked
__device__ __forceinline__ uint4 pk8_bf16(float4 f0, float4 f1, unsigned amask) {
    uint4 r;
    r.x = pk_bf16(f0.x, f0.y) & amask;
    r.y = pk_bf16(f0.z, f0.w) & amask;
    r.z = pk_bf16(f1.x, f1.y) & amask;
    r.w = pk_bf16(f1.z, f1.w) & amask;
    return r;
}

union FragU { v16bf v; uint4 u[2]; };

// A fragment, 16x32 bf16 (M x K), source row-major with element stride `stride`.
// Contiguous-k layout used here (matches frag_b): lanes 0-15 row M=lane, K=0..15; lanes 16-31 K=16..31.
__device__ __forceinline__ v16bf frag_a(const unsigned short* base, int stride) {
    const int lane = threadIdx.x & 31;
    const int row  = lane & 15;
    const int koff = (lane >> 4) << 4;            // 0 or 16
    const unsigned short* p = base + row * stride + koff;
    FragU f;
    f.u[0] = *(const uint4*)(p);                  // K = koff .. koff+7
    f.u[1] = *(const uint4*)(p + 8);              // K = koff+8 .. koff+15
    return f.v;
}

// B fragment, 32x16 bf16 (K x N) where B[k][n] = src[n][k] (src row-major, K contiguous).
// Contiguous-k layout used here (matches frag_a): lane = column n (mod 16); lanes 0-15 K=0..15, lanes 16-31 K=16..31.
__device__ __forceinline__ v16bf frag_b(const unsigned short* base, int stride) {
    const int lane = threadIdx.x & 31;
    const int col  = lane & 15;
    const int koff = (lane >> 4) << 4;            // 0 or 16
    const unsigned short* p = base + col * stride + koff;
    FragU f;
    f.u[0] = *(const uint4*)(p);
    f.u[1] = *(const uint4*)(p + 8);
    return f.v;
}

#define WMMA_BF16(A, B, C) \
    __builtin_amdgcn_wmma_f32_16x16x32_bf16(false, (A), false, (B), (short)0, (C), false, false)

// ---------------- weight conversion: 8 fp32 -> 8 bf16 per thread ----------------
__global__ void lwa_convert_weights(const float* __restrict__ wqkv,
                                    const float* __restrict__ wout,
                                    unsigned short* __restrict__ wqkv_b,
                                    unsigned short* __restrict__ wout_b) {
    const int idx = (blockIdx.x * 256 + threadIdx.x) * 8;   // grid covers NQKV*EMB/8
    {
        const float4 f0 = *(const float4*)(wqkv + idx);
        const float4 f1 = *(const float4*)(wqkv + idx + 4);
        const uint4 r = pk8_bf16(f0, f1, 0xffffffffu);
        *(uint4*)(wqkv_b + idx) = r;
    }
    if (idx < EMB * EMB) {
        const float4 f0 = *(const float4*)(wout + idx);
        const float4 f1 = *(const float4*)(wout + idx + 4);
        const uint4 r = pk8_bf16(f0, f1, 0xffffffffu);
        *(uint4*)(wout_b + idx) = r;
    }
}

// ---------------- QKV GEMM: qkv[32768][3072] = pad(x)@wqkv^T + b, bf16 out ----
// Workgroup tile 128(M) x 256(N), BK=32, ping-pong LDS, one barrier/iter.
// 8 waves in 2x4 grid, 64x64 per wave (16 accumulators).
// A read directly from fp32 x: safe clamped pointer + dword mask (no exec branch).
__global__ __launch_bounds__(256)
void lwa_qkv_gemm(const float* __restrict__ X,             // [8][4000][1024] f32
                  const unsigned short* __restrict__ Bw,   // [3072][1024]  bf16
                  const float* __restrict__ bias,          // [3072]
                  unsigned short* __restrict__ C) {        // [32768][3072] bf16
    __shared__ __align__(16) unsigned short As[2][128 * 32];   // 16 KB
    __shared__ __align__(16) unsigned short Bs[2][256 * 32];   // 32 KB
    const int tn = blockIdx.x, tm = blockIdx.y;
    const int tid = threadIdx.x;
    const int w  = tid >> 5;
    const int wm = (w & 1) * 64;       // wave M base within tile
    const int wn = (w >> 1) * 64;      // wave N base within tile
    // A staging: 2 threads per row, 16 fp32 each
    const int ar = tid >> 1;
    const int ac = (tid & 1) << 4;
    const int arow = tm * 128 + ar;
    const int ab   = arow >> 12;               // batch
    const int as   = arow & (SEQP - 1);        // seq within padded batch
    const bool a_ok = (as < SEQ);
    const unsigned amask = a_ok ? 0xffffffffu : 0u;            // zero pad rows
    const float* Ag = X + ((size_t)(ab * SEQ + (a_ok ? as : 0)) * EMB + ac);
    // B staging: 1 thread per row, 32 bf16 each
    const unsigned short* Bg = Bw + (size_t)(tn * 256 + tid) * EMB;

    v8f acc[4][4] = {};

    // ---- prologue fetch (k = 0) ----
    uint4 apk0, apk1;                  // packed masked bf16 A (16 elems)
    uint4 bq0, bq1, bq2, bq3;          // B (32 bf16)
    {
        const float4 f0 = *(const float4*)(Ag);
        const float4 f1 = *(const float4*)(Ag + 4);
        const float4 f2 = *(const float4*)(Ag + 8);
        const float4 f3 = *(const float4*)(Ag + 12);
        apk0 = pk8_bf16(f0, f1, amask);
        apk1 = pk8_bf16(f2, f3, amask);
        bq0 = *(const uint4*)(Bg);
        bq1 = *(const uint4*)(Bg + 8);
        bq2 = *(const uint4*)(Bg + 16);
        bq3 = *(const uint4*)(Bg + 24);
    }

    int p = 0;
    for (int k0 = 0; k0 < EMB; k0 += 32, p ^= 1) {
        // stage current K-slab into LDS buffer p
        *(uint4*)(&As[p][ar * 32 + ac])     = apk0;
        *(uint4*)(&As[p][ar * 32 + ac + 8]) = apk1;
        *(uint4*)(&Bs[p][tid * 32])      = bq0;
        *(uint4*)(&Bs[p][tid * 32 + 8])  = bq1;
        *(uint4*)(&Bs[p][tid * 32 + 16]) = bq2;
        *(uint4*)(&Bs[p][tid * 32 + 24]) = bq3;
        __syncthreads();
        // issue next slab's global fetch (clamped on final iteration)
        const int kn = (k0 + 32 < EMB) ? (k0 + 32) : k0;
        const float4 f0 = *(const float4*)(Ag + kn);
        const float4 f1 = *(const float4*)(Ag + kn + 4);
        const float4 f2 = *(const float4*)(Ag + kn + 8);
        const float4 f3 = *(const float4*)(Ag + kn + 12);
        bq0 = *(const uint4*)(Bg + kn);
        bq1 = *(const uint4*)(Bg + kn + 8);
        bq2 = *(const uint4*)(Bg + kn + 16);
        bq3 = *(const uint4*)(Bg + kn + 24);
        // 8 fragment loads feed 16 WMMAs (global fetches in flight above)
        v16bf bf[4];
#pragma unroll
        for (int ni = 0; ni < 4; ++ni) bf[ni] = frag_b(&Bs[p][(wn + ni * 16) * 32], 32);
#pragma unroll
        for (int mi = 0; mi < 4; ++mi) {
            const v16bf af = frag_a(&As[p][(wm + mi * 16) * 32], 32);
#pragma unroll
            for (int ni = 0; ni < 4; ++ni)
                acc[mi][ni] = WMMA_BF16(af, bf[ni], acc[mi][ni]);
        }
        // pack next A (3 VALU per pair via v_perm_b32; scheduled after WMMAs)
        apk0 = pk8_bf16(f0, f1, amask);
        apk1 = pk8_bf16(f2, f3, amask);
    }
    const int lane = tid & 31;
    const int rsub = (lane >> 4) << 3;
    const int cc   = lane & 15;
#pragma unroll
    for (int mi = 0; mi < 4; ++mi)
#pragma unroll
        for (int ni = 0; ni < 4; ++ni)
#pragma unroll
            for (int i = 0; i < 8; ++i) {
                const int row = tm * 128 + wm + mi * 16 + i + rsub;
                const int col = tn * 256 + wn + ni * 16 + cc;
                C[(size_t)row * NQKV + col] = f2bf(acc[mi][ni][i] + bias[col]);
            }
}

// ---------------- per-window attention: one WG per (head, window, batch) ----------------
__global__ __launch_bounds__(256)
void lwa_attention(const unsigned short* __restrict__ qkv,   // [32768][3072] bf16
                   unsigned short* __restrict__ ob) {        // [32768][1024] bf16
    __shared__ __align__(16) unsigned short qs[128 * 64];
    __shared__ __align__(16) unsigned short ks[128 * 64];
    __shared__ __align__(16) unsigned short vt[64 * 128];    // transposed: vt[d][k]
    __shared__ __align__(16) unsigned short ps[128 * 128];   // probabilities bf16

    const int h = blockIdx.x, n = blockIdx.y, b = blockIdx.z;
    const int tid  = threadIdx.x;
    const int w    = tid >> 5;
    const int lane = tid & 31;
    const size_t rowbase = (size_t)(b * NWIN + n) * 128;

    // ---- stage q, k, v(T) tiles into LDS: 2 threads per row, 32 elems each ----
    {
        const int lr = tid >> 1;
        const int lc = (tid & 1) << 5;
        const unsigned short* src = qkv + (rowbase + lr) * NQKV + h * DHEAD + lc;
        uint4 t0 = *(const uint4*)(src);
        uint4 t1 = *(const uint4*)(src + 8);
        uint4 t2 = *(const uint4*)(src + 16);
        uint4 t3 = *(const uint4*)(src + 24);
        *(uint4*)(qs + lr * 64 + lc)      = t0;
        *(uint4*)(qs + lr * 64 + lc + 8)  = t1;
        *(uint4*)(qs + lr * 64 + lc + 16) = t2;
        *(uint4*)(qs + lr * 64 + lc + 24) = t3;
        t0 = *(const uint4*)(src + EMB);
        t1 = *(const uint4*)(src + EMB + 8);
        t2 = *(const uint4*)(src + EMB + 16);
        t3 = *(const uint4*)(src + EMB + 24);
        *(uint4*)(ks + lr * 64 + lc)      = t0;
        *(uint4*)(ks + lr * 64 + lc + 8)  = t1;
        *(uint4*)(ks + lr * 64 + lc + 16) = t2;
        *(uint4*)(ks + lr * 64 + lc + 24) = t3;
        union { uint4 u[4]; unsigned short s[32]; } vb;
        vb.u[0] = *(const uint4*)(src + 2 * EMB);
        vb.u[1] = *(const uint4*)(src + 2 * EMB + 8);
        vb.u[2] = *(const uint4*)(src + 2 * EMB + 16);
        vb.u[3] = *(const uint4*)(src + 2 * EMB + 24);
#pragma unroll
        for (int j = 0; j < 32; ++j) vt[(lc + j) * 128 + lr] = vb.s[j];
    }
    __syncthreads();

    // ---- scores: wave w owns rows w*16..w*16+15, all 128 cols ----
    v8f sacc[8] = {};
#pragma unroll
    for (int kk = 0; kk < DHEAD; kk += 32) {
        const v16bf af = frag_a(qs + (w * 16) * 64 + kk, 64);
#pragma unroll
        for (int ni = 0; ni < 8; ++ni) {
            const v16bf bf = frag_b(ks + (ni * 16) * 64 + kk, 64);
            sacc[ni] = WMMA_BF16(af, bf, sacc[ni]);
        }
    }

    // ---- row softmax in registers (row = w*16 + i + 8*(lane>=16)) ----
    const float scale = 0.125f;   // 1/sqrt(64)
#pragma unroll
    for (int i = 0; i < 8; ++i) {
        float m = -3.0e38f;
#pragma unroll
        for (int ni = 0; ni < 8; ++ni) {
            sacc[ni][i] *= scale;
            m = fmaxf(m, sacc[ni][i]);
        }
#pragma unroll
        for (int off = 1; off < 16; off <<= 1) m = fmaxf(m, __shfl_xor(m, off, 32));
        float e[8], sum = 0.0f;
#pragma unroll
        for (int ni = 0; ni < 8; ++ni) { e[ni] = __expf(sacc[ni][i] - m); sum += e[ni]; }
#pragma unroll
        for (int off = 1; off < 16; off <<= 1) sum += __shfl_xor(sum, off, 32);
        const float inv = 1.0f / sum;
        const int row = w * 16 + i + ((lane >> 4) << 3);
#pragma unroll
        for (int ni = 0; ni < 8; ++ni)
            ps[row * 128 + ni * 16 + (lane & 15)] = f2bf(e[ni] * inv);
    }
    __syncthreads();

    // ---- O = P @ V : wave strip 16 x 64, K = 128 ----
    v8f oacc[4] = {};
#pragma unroll
    for (int kk = 0; kk < 128; kk += 32) {
        const v16bf af = frag_a(ps + (w * 16) * 128 + kk, 128);
#pragma unroll
        for (int ni = 0; ni < 4; ++ni) {
            const v16bf bf = frag_b(vt + (ni * 16) * 128 + kk, 128);
            oacc[ni] = WMMA_BF16(af, bf, oacc[ni]);
        }
    }
#pragma unroll
    for (int ni = 0; ni < 4; ++ni)
#pragma unroll
        for (int i = 0; i < 8; ++i) {
            const int row = w * 16 + i + ((lane >> 4) << 3);
            const int col = h * DHEAD + ni * 16 + (lane & 15);
            ob[(rowbase + row) * EMB + col] = f2bf(oacc[ni][i]);
        }
}

// ---------------- output GEMM: out = ob @ w_out^T + b_out, fp32, drop padded rows ----
// Same 128x256 tiling, ping-pong LDS, bf16 A from attention output buffer.
__global__ __launch_bounds__(256)
void lwa_out_gemm(const unsigned short* __restrict__ A,    // [32768][1024] bf16
                  const unsigned short* __restrict__ Bw,   // [1024][1024]  bf16
                  const float* __restrict__ bias,          // [1024]
                  float* __restrict__ out) {               // [8][4000][1024] f32
    __shared__ __align__(16) unsigned short As[2][128 * 32];
    __shared__ __align__(16) unsigned short Bs[2][256 * 32];
    const int tn = blockIdx.x, tm = blockIdx.y;
    const int tid = threadIdx.x;
    const int w  = tid >> 5;
    const int wm = (w & 1) * 64;
    const int wn = (w >> 1) * 64;
    const int ar = tid >> 1;
    const int ac = (tid & 1) << 4;

    v8f acc[4][4] = {};
    const unsigned short* Ag = A  + (size_t)(tm * 128 + ar) * EMB + ac;
    const unsigned short* Bg = Bw + (size_t)(tn * 256 + tid) * EMB;

    uint4 aq0 = *(const uint4*)(Ag);
    uint4 aq1 = *(const uint4*)(Ag + 8);
    uint4 bq0 = *(const uint4*)(Bg);
    uint4 bq1 = *(const uint4*)(Bg + 8);
    uint4 bq2 = *(const uint4*)(Bg + 16);
    uint4 bq3 = *(const uint4*)(Bg + 24);

    int p = 0;
    for (int k0 = 0; k0 < EMB; k0 += 32, p ^= 1) {
        *(uint4*)(&As[p][ar * 32 + ac])     = aq0;
        *(uint4*)(&As[p][ar * 32 + ac + 8]) = aq1;
        *(uint4*)(&Bs[p][tid * 32])      = bq0;
        *(uint4*)(&Bs[p][tid * 32 + 8])  = bq1;
        *(uint4*)(&Bs[p][tid * 32 + 16]) = bq2;
        *(uint4*)(&Bs[p][tid * 32 + 24]) = bq3;
        __syncthreads();
        const int kn = (k0 + 32 < EMB) ? (k0 + 32) : k0;
        aq0 = *(const uint4*)(Ag + kn);
        aq1 = *(const uint4*)(Ag + kn + 8);
        bq0 = *(const uint4*)(Bg + kn);
        bq1 = *(const uint4*)(Bg + kn + 8);
        bq2 = *(const uint4*)(Bg + kn + 16);
        bq3 = *(const uint4*)(Bg + kn + 24);
        v16bf bf[4];
#pragma unroll
        for (int ni = 0; ni < 4; ++ni) bf[ni] = frag_b(&Bs[p][(wn + ni * 16) * 32], 32);
#pragma unroll
        for (int mi = 0; mi < 4; ++mi) {
            const v16bf af = frag_a(&As[p][(wm + mi * 16) * 32], 32);
#pragma unroll
            for (int ni = 0; ni < 4; ++ni)
                acc[mi][ni] = WMMA_BF16(af, bf[ni], acc[mi][ni]);
        }
    }
    const int lane = tid & 31;
    const int rsub = (lane >> 4) << 3;
    const int cc   = lane & 15;
#pragma unroll
    for (int mi = 0; mi < 4; ++mi)
#pragma unroll
        for (int ni = 0; ni < 4; ++ni)
#pragma unroll
            for (int i = 0; i < 8; ++i) {
                const int row = tm * 128 + wm + mi * 16 + i + rsub;
                const int col = tn * 256 + wn + ni * 16 + cc;
                const int bb = row >> 12;
                const int s  = row & (SEQP - 1);
                if (s < SEQ)
                    out[(size_t)(bb * SEQ + s) * EMB + col] = acc[mi][ni][i] + bias[col];
            }
}

// ---------------- launcher ----------------
extern "C" void kernel_launch(void* const* d_in, const int* in_sizes, int n_in,
                              void* d_out, int out_size, void* d_ws, size_t ws_size,
                              hipStream_t stream) {
    const float* x     = (const float*)d_in[0];
    const float* w_qkv = (const float*)d_in[1];
    const float* b_qkv = (const float*)d_in[2];
    const float* w_out = (const float*)d_in[3];
    const float* b_out = (const float*)d_in[4];
    float* out = (float*)d_out;

    unsigned short* ws     = (unsigned short*)d_ws;
    unsigned short* wqkv_b = ws;                                       // 3,145,728 elems
    unsigned short* wout_b = wqkv_b + (size_t)NQKV * EMB;              // 1,048,576
    unsigned short* qkvb   = wout_b + (size_t)EMB * EMB;               // 100,663,296
    unsigned short* obuf   = qkvb + (size_t)MROWS * NQKV;              // 33,554,432

    lwa_convert_weights<<<(NQKV * EMB) / (256 * 8), 256, 0, stream>>>(w_qkv, w_out, wqkv_b, wout_b);
    lwa_qkv_gemm<<<dim3(NQKV / 256, MROWS / 128), 256, 0, stream>>>(x, wqkv_b, b_qkv, qkvb);
    lwa_attention<<<dim3(HEADS, NWIN, BATCH), 256, 0, stream>>>(qkvb, obuf);
    lwa_out_gemm<<<dim3(EMB / 256, MROWS / 128), 256, 0, stream>>>(obuf, wout_b, b_out, out);
}
